// VectorQuantizer_60962765800287
// MI455X (gfx1250) — matrix operand order repaired and checked by reading, not compiled, not run on hardware
//
#include <hip/hip_runtime.h>
#include <hip/hip_bf16.h>

// ---------------------------------------------------------------------------
// VQ-VAE vector quantizer for MI455X (gfx1250, wave32, WMMA, async-LDS,
// double-buffered producer/consumer pipeline)
//   x:          [32,32,32,256] f32  -> flat [N=32768, D=256]
//   embeddings: [D=256, K=1024] f32
//   out:        [N*D] quantized_st f32  ++  [1] loss f32
// ---------------------------------------------------------------------------

typedef __attribute__((ext_vector_type(16))) __bf16          v16bf;
typedef __attribute__((ext_vector_type(8)))  float           v8f;
typedef __attribute__((ext_vector_type(16))) unsigned short  v16u;
typedef __attribute__((ext_vector_type(4)))  int             v4i;

#define VQ_D 256
#define VQ_K 1024
#define VQ_N 32768

// LDS staging: 64 codes per chunk, row stride padded 512B -> 528B so that
// lane l of each 16-lane half hits bank set {4l..4l+3} (+8 for the upper
// half): each of the 64 banks is requested exactly twice per ds_load_b128,
// the 512B/instr minimum -> conflict-free.
#define VQ_CHUNK      64
#define VQ_NCHUNK     (VQ_K / VQ_CHUNK)
#define VQ_ROW_BYTES  512          // 256 * bf16
#define VQ_ROW_STRIDE 528          // +16B pad

#if defined(__gfx1250__) && __has_builtin(__builtin_amdgcn_global_load_async_to_lds_b128)
#define VQ_ASYNC_LDS 1
#endif

static __device__ __forceinline__ unsigned short f2bf_rne(float f) {
    unsigned u = __builtin_bit_cast(unsigned, f);
    u += 0x7FFFu + ((u >> 16) & 1u);          // round-to-nearest-even
    return (unsigned short)(u >> 16);
}

struct Q2 { uint4 a, b; };                    // 32B container for a B fragment

// ---------------------------------------------------------------------------
// Prep: transpose codebook to [K, D] (f32 + bf16) and compute ||e_k||^2.
// ---------------------------------------------------------------------------
__global__ __launch_bounds__(256) void vq_prep_kernel(
        const float* __restrict__ emb,          // [D, K]
        unsigned short* __restrict__ Etb,       // [K, D] bf16 bits
        float* __restrict__ Etf,                // [K, D] f32
        float* __restrict__ enorm)              // [K]
{
    const int k = blockIdx.x;
    const int d = threadIdx.x;
    const float v = emb[(size_t)d * VQ_K + k];
    Etf[(size_t)k * VQ_D + d] = v;
    Etb[(size_t)k * VQ_D + d] = f2bf_rne(v);

    __shared__ float red[256];
    red[d] = v * v;
    __syncthreads();
    #pragma unroll
    for (int s = 128; s > 0; s >>= 1) {
        if (d < s) red[d] += red[d + s];
        __syncthreads();
    }
    if (d == 0) enorm[k] = red[0];
}

// ---------------------------------------------------------------------------
// GEMM + fused argmin. One wave per 16-row block of x; 8 waves / workgroup.
// A (16x256 of x, bf16) lives in 64 VGPRs, reused across all 1024 codes.
// B is staged per 64-code chunk into LDS via CDNA5 async copies, shared by
// all 8 waves, and double-buffered: chunk i+1 streams in (ASYNCcnt DMA)
// while chunk i's 32 WMMAs execute. ks-outer / t4-inner ordering keeps 4
// independent accumulators in flight so ds_loads pipeline under WMMAs.
// ---------------------------------------------------------------------------
__global__ __launch_bounds__(256) void vq_argmin_kernel(
        const float* __restrict__ X,            // [N, D]
        const unsigned short* __restrict__ Etb, // [K, D] bf16 bits
        const float* __restrict__ enorm,        // [K]
        int* __restrict__ idxOut)               // [N]
{
    __shared__ __align__(16) unsigned char sEt[2][VQ_CHUNK * VQ_ROW_STRIDE];

    const int lane    = threadIdx.x & 31;
    const int wave    = threadIdx.x >> 5;
    const int rowblk  = blockIdx.x * 8 + wave;
    const int row0    = rowblk * 16;
    const int halfsel = lane >> 4;              // 0: lanes 0-15, 1: lanes 16-31
    const int l15     = lane & 15;

    // Cooperative copy role: thread t -> row r = t/4, quarter q = t%4 (128B).
    const int cp_r = threadIdx.x >> 2;
    const int cp_q = threadIdx.x & 3;

    // Issue the 32KB fill of one chunk into LDS buffer `buf`.
    auto issue_chunk = [&](int chunk, int buf) {
        const unsigned char* gsrc = (const unsigned char*)Etb
            + (size_t)chunk * VQ_CHUNK * VQ_ROW_BYTES
            + cp_r * VQ_ROW_BYTES + cp_q * 128;
        unsigned char* ldst = sEt[buf] + cp_r * VQ_ROW_STRIDE + cp_q * 128;
#if VQ_ASYNC_LDS
        #pragma unroll
        for (int j = 0; j < 8; ++j) {
            __builtin_amdgcn_global_load_async_to_lds_b128(
                (__attribute__((address_space(1))) v4i*)(gsrc + j * 16),
                (__attribute__((address_space(3))) v4i*)(ldst + j * 16),
                0, 0);
        }
#else
        #pragma unroll
        for (int j = 0; j < 8; ++j)
            *(uint4*)(ldst + j * 16) = *(const uint4*)(gsrc + j * 16);
#endif
    };

    // ---- Kick off chunk 0 DMA, then overlap it with the A-fragment load.
    issue_chunk(0, 0);

    // ---- Load A fragments (contiguous k order, same as B):
    //      lane (l15) = row; element e -> K = 32*ks + 16*halfsel + e
    v16bf A[8];
    {
        const float* xr = X + (size_t)(row0 + l15) * VQ_D;
        #pragma unroll
        for (int ks = 0; ks < 8; ++ks) {
            const int kb = ks * 32 + halfsel * 16;
            v16u uv;
            #pragma unroll
            for (int e = 0; e < 8; ++e)  uv[e] = f2bf_rne(xr[kb + e]);
            #pragma unroll
            for (int e = 8; e < 16; ++e) uv[e] = f2bf_rne(xr[kb + e]);
            A[ks] = __builtin_bit_cast(v16bf, uv);
        }
    }

    float minv[8];
    int   mini[8];
    #pragma unroll
    for (int i = 0; i < 8; ++i) { minv[i] = 3.4e38f; mini[i] = 0; }

    for (int chunk = 0; chunk < VQ_NCHUNK; ++chunk) {
        const int cur = chunk & 1;

#if VQ_ASYNC_LDS
        // Only this chunk's 8 loads are outstanding; they had all of the
        // previous chunk's compute to finish in the background.
        asm volatile("s_wait_asynccnt 0x0" ::: "memory");
#endif
        // All waves: own DMA done AND previous chunk's compute done.
        // (A wave reaching issue_chunk below therefore cannot clobber a
        //  buffer any wave is still reading.)
        __syncthreads();

        // Branchless next-chunk issue: the wrap-around fill on the final
        // iteration lands in the dead buffer (never read; s_endpgm's
        // implicit wait-idle drains it).
        issue_chunk((chunk + 1) & (VQ_NCHUNK - 1), cur ^ 1);

        const int cbase = chunk * VQ_CHUNK;

        // Hoist the 4 e-norm loads so the epilogue doesn't stall on them.
        float en[4];
        #pragma unroll
        for (int t4 = 0; t4 < 4; ++t4) en[t4] = enorm[cbase + t4 * 16 + l15];

        // ---- 4 column tiles of 16 codes, B fragments from LDS.
        //      ks outer / t4 inner: 4 independent accumulator chains.
        v8f acc[4] = {};
        const unsigned char* base = sEt[cur]
            + (size_t)l15 * VQ_ROW_STRIDE + halfsel * 32;
        #pragma unroll
        for (int ks = 0; ks < 8; ++ks) {
            #pragma unroll
            for (int t4 = 0; t4 < 4; ++t4) {
                // B (32x16, 16-bit): col = l15, element e -> K = 16*halfsel + e
                const unsigned char* bp = base
                    + t4 * 16 * VQ_ROW_STRIDE + ks * 64;
                Q2 q;
                q.a = *(const uint4*)(bp);
                q.b = *(const uint4*)(bp + 16);
                v16bf B = __builtin_bit_cast(v16bf, q);
                acc[t4] = __builtin_amdgcn_wmma_f32_16x16x32_bf16(
                        false, A[ks], false, B, (short)0, acc[t4], false, false);
            }
        }

        // dist = ||e||^2 - 2*sim  (||x||^2 row-constant -> argmin invariant)
        #pragma unroll
        for (int t4 = 0; t4 < 4; ++t4) {
            const int mycol = cbase + t4 * 16 + l15;
            #pragma unroll
            for (int i = 0; i < 8; ++i) {
                const float dist = en[t4] - 2.0f * acc[t4][i];
                if (dist < minv[i]) { minv[i] = dist; mini[i] = mycol; }
            }
        }
    }

    // ---- Reduce (min,idx) across the 16 lanes holding each row.
    //      C/D layout: VGPR i -> row i (lanes 0-15) / row 8+i (lanes 16-31).
    #pragma unroll
    for (int i = 0; i < 8; ++i) {
        float v  = minv[i];
        int   ix = mini[i];
        #pragma unroll
        for (int off = 8; off > 0; off >>= 1) {
            const float ov = __shfl_xor(v, off, 16);
            const int   oi = __shfl_xor(ix, off, 16);
            if (ov < v || (ov == v && oi < ix)) { v = ov; ix = oi; }
        }
        if (l15 == 0) idxOut[row0 + halfsel * 8 + i] = ix;
    }
}

// ---------------------------------------------------------------------------
// Gather + straight-through output + per-row squared-error partials.
// ---------------------------------------------------------------------------
__global__ __launch_bounds__(256) void vq_gather_kernel(
        const float* __restrict__ X,
        const float* __restrict__ Etf,          // [K, D] f32
        const int* __restrict__ idx,
        float* __restrict__ out,                // [N*D]
        float* __restrict__ partials)           // [N]
{
    const int n = blockIdx.x;
    const int d = threadIdx.x;
    const int k = idx[n];
    const float q = Etf[(size_t)k * VQ_D + d];
    const float x = X[(size_t)n * VQ_D + d];
    const float t = q - x;                      // stop_grad(quantized - x)
    out[(size_t)n * VQ_D + d] = x + t;          // straight-through estimator

    __shared__ float red[256];
    red[d] = t * t;
    __syncthreads();
    #pragma unroll
    for (int s = 128; s > 0; s >>= 1) {
        if (d < s) red[d] += red[d + s];
        __syncthreads();
    }
    if (d == 0) partials[n] = red[0];
}

// ---------------------------------------------------------------------------
// Deterministic final reduction: loss = 1.25 * sum / (N*D).
// ---------------------------------------------------------------------------
__global__ __launch_bounds__(256) void vq_loss_kernel(
        const float* __restrict__ partials,
        float* __restrict__ out)
{
    const int tid = threadIdx.x;
    float s = 0.f;
    for (int i = tid; i < VQ_N; i += 256) s += partials[i];   // fixed order

    __shared__ float red[256];
    red[tid] = s;
    __syncthreads();
    #pragma unroll
    for (int st = 128; st > 0; st >>= 1) {
        if (tid < st) red[tid] += red[tid + st];
        __syncthreads();
    }
    if (tid == 0)
        out[(size_t)VQ_N * VQ_D] = 1.25f * red[0] / (float)(VQ_N * VQ_D);
}

// ---------------------------------------------------------------------------
extern "C" void kernel_launch(void* const* d_in, const int* in_sizes, int n_in,
                              void* d_out, int out_size, void* d_ws, size_t ws_size,
                              hipStream_t stream)
{
    const float* x   = (const float*)d_in[0];   // [N, D]
    const float* emb = (const float*)d_in[1];   // [D, K]
    float* out = (float*)d_out;                 // [N*D + 1]
    char*  ws  = (char*)d_ws;

    unsigned short* Etb     = (unsigned short*)(ws);            // 512 KB
    float*          Etf     = (float*)(ws + 524288);            // 1 MB
    float*          enorm   = (float*)(ws + 1572864);           // 4 KB
    int*            idx     = (int*)  (ws + 1576960);           // 128 KB
    float*          partial = (float*)(ws + 1708032);           // 128 KB

    hipLaunchKernelGGL(vq_prep_kernel,   dim3(VQ_K),        dim3(256), 0, stream,
                       emb, Etb, Etf, enorm);
    hipLaunchKernelGGL(vq_argmin_kernel, dim3(VQ_N / 128),  dim3(256), 0, stream,
                       x, Etb, enorm, idx);
    hipLaunchKernelGGL(vq_gather_kernel, dim3(VQ_N),        dim3(256), 0, stream,
                       x, Etf, idx, out, partial);
    hipLaunchKernelGGL(vq_loss_kernel,   dim3(1),           dim3(256), 0, stream,
                       partial, out);
}
